// NeuronAttention_40450001994295
// MI455X (gfx1250) — matrix operand order repaired and checked by reading, not compiled, not run on hardware
//
#include <hip/hip_runtime.h>

// Problem constants (match reference)
#define B_  2
#define S_  1024
#define D_  2048
#define H_  16
#define HD_ 128
#define CS_ 2048
#define NQK (H_*HD_)   // 2048

typedef __attribute__((ext_vector_type(16))) __bf16 bf16x16;
typedef __attribute__((ext_vector_type(8)))  float  f32x8;

union Frag16 { bf16x16 v; unsigned short u[16]; };

__device__ __forceinline__ unsigned short f32_bf16(float f) {
  unsigned int x = __float_as_uint(f);
  x += 0x7FFFu + ((x >> 16) & 1u);           // round-to-nearest-even
  return (unsigned short)(x >> 16);
}

__device__ __forceinline__ f32x8 zero8() {
  f32x8 z = {0.f,0.f,0.f,0.f,0.f,0.f,0.f,0.f};
  return z;
}

// CDNA5 async DMA: copy 16 bytes/lane from global (SADDR base + 32-bit voffset)
// into LDS at per-lane address lds_off. Tracked by ASYNCcnt.
__device__ __forceinline__ void async_copy16(unsigned lds_off, const void* base, unsigned voff) {
  asm volatile("global_load_async_to_lds_b128 %0, %1, %2"
               : : "v"(lds_off), "v"(voff), "s"(base) : "memory");
}

// ---------------------------------------------------------------------------
// Elementwise f32 -> bf16
// ---------------------------------------------------------------------------
__global__ void f32_to_bf16_vec(const float* __restrict__ src,
                                unsigned short* __restrict__ dst, int n)
{
  int i = blockIdx.x*blockDim.x + threadIdx.x;
  if (i < n) dst[i] = f32_bf16(src[i]);
}

// ---------------------------------------------------------------------------
// Transpose-convert: W f32 [k][n] (2048x2048) -> Wt bf16 [n][k]
// ---------------------------------------------------------------------------
__global__ __launch_bounds__(256)
void transpose_f32_to_bf16(const float* __restrict__ W, unsigned short* __restrict__ Wt)
{
  __shared__ unsigned short tile[32][33];
  const int tx = threadIdx.x & 31;
  const int ty = threadIdx.x >> 5;   // 0..7
  const int k0 = blockIdx.x * 32;
  const int n0 = blockIdx.y * 32;
  #pragma unroll
  for (int r = 0; r < 4; ++r) {
    int k = ty + r*8;
    tile[k][tx] = f32_bf16(W[(size_t)(k0 + k)*2048 + n0 + tx]);
  }
  __syncthreads();
  #pragma unroll
  for (int r = 0; r < 4; ++r) {
    int n = ty + r*8;
    Wt[(size_t)(n0 + n)*2048 + k0 + tx] = tile[tx][n];
  }
}

// ---------------------------------------------------------------------------
// 2048^3 GEMM, all-bf16 operands:  C = A(bf16, MxK row-major) @ Bt(bf16, NxK)^T
// Block 128x128, 8 waves, each 32x64 (2x4 wmma accums). Double-buffered LDS
// filled by GLOBAL_LOAD_ASYNC_TO_LDS_B128 (4 async instrs / wave / tile).
// dstF (f32) and dstB (bf16) stores, each with a layout code:
//   0: row-major [row*2048 + col]
//   1: head/cache layout ((b*H+h)*Tdim + s)*HD + hd
//   2: transposed head layout ((b*H+h)*HD + hd)*CS + s
// ---------------------------------------------------------------------------
__global__ __launch_bounds__(256)
void gemm_bf16(const unsigned short* __restrict__ A,
               const unsigned short* __restrict__ Bt,
               const float* __restrict__ bias,
               float* __restrict__ dstF, int layoutF,
               unsigned short* __restrict__ dstB, int layoutB,
               int Tdim)
{
  __shared__ unsigned short sA[2][128][32];   // [buf][m][k]
  __shared__ unsigned short sB[2][128][32];   // [buf][n][k]
  const int tid   = threadIdx.x;
  const int lane  = tid & 31;
  const int wave  = tid >> 5;
  const int waveM = wave >> 1;   // 0..3
  const int waveN = wave & 1;    // 0..1
  const int blockM = blockIdx.x * 128;
  const int blockN = blockIdx.y * 128;

  const int lm = lane & 15;
  const int kg = (lane >> 4) * 8;    // K group (0 / 8), unused
  const int kh = (lane >> 4) * 16;   // B-frag K half  (0 / 16)

  const unsigned sA_base = (unsigned)(size_t)&sA[0][0][0];
  const unsigned sB_base = (unsigned)(size_t)&sB[0][0][0];

  // stage one 128x32 bf16 tile pair (A + B) into LDS buffer `buf`
  auto stage = [&](int k0, int buf) {
    #pragma unroll
    for (int i = 0; i < 2; ++i) {
      int c = tid + i*256;                 // 0..511 : 128 rows x 4 x 16B
      int row = c >> 2, q16 = c & 3;
      unsigned lds  = sA_base + (unsigned)buf*8192u + (unsigned)row*64u + (unsigned)q16*16u;
      unsigned voff = (unsigned)((((size_t)(blockM + row))*2048 + k0 + q16*8) * 2);
      async_copy16(lds, A, voff);
    }
    #pragma unroll
    for (int i = 0; i < 2; ++i) {
      int c = tid + i*256;
      int row = c >> 2, q16 = c & 3;
      unsigned lds  = sB_base + (unsigned)buf*8192u + (unsigned)row*64u + (unsigned)q16*16u;
      unsigned voff = (unsigned)((((size_t)(blockN + row))*2048 + k0 + q16*8) * 2);
      async_copy16(lds, Bt, voff);
    }
  };

  f32x8 acc[2][4];
  #pragma unroll
  for (int i = 0; i < 2; ++i)
    #pragma unroll
    for (int j = 0; j < 4; ++j) acc[i][j] = zero8();

  stage(0, 0);
  for (int k0 = 0; k0 < 2048; k0 += 32) {
    const int cur = (k0 >> 5) & 1;
    if (k0 + 32 < 2048) {
      stage(k0 + 32, cur ^ 1);
      asm volatile("s_wait_asynccnt 0x4" ::: "memory");  // drain current tile
    } else {
      asm volatile("s_wait_asynccnt 0x0" ::: "memory");
    }
    __syncthreads();

    Frag16 a[2], b[4];
    #pragma unroll
    for (int t = 0; t < 2; ++t) {
      const unsigned short* ap = &sA[cur][waveM*32 + t*16 + lm][0];
      #pragma unroll
      for (int p = 0; p < 8; ++p) {
        int kk = kh + 2*p;
        a[t].u[2*p]   = ap[kk];
        a[t].u[2*p+1] = ap[kk+1];
      }
    }
    #pragma unroll
    for (int t = 0; t < 4; ++t) {
      const unsigned short* bp = &sB[cur][waveN*64 + t*16 + lm][0];
      #pragma unroll
      for (int p = 0; p < 8; ++p) {
        b[t].u[2*p]   = bp[kh + 2*p];
        b[t].u[2*p+1] = bp[kh + 2*p + 1];
      }
    }
    #pragma unroll
    for (int i = 0; i < 2; ++i)
      #pragma unroll
      for (int j = 0; j < 4; ++j)
        acc[i][j] = __builtin_amdgcn_wmma_f32_16x16x32_bf16(
            false, a[i].v, false, b[j].v, (short)0, acc[i][j], false, false);
    __syncthreads();   // protect read buffer from next iteration's DMA writes
  }

  const int mhalf = (lane >> 4) * 8;
  #pragma unroll
  for (int j = 0; j < 4; ++j) {
    const int colG = blockN + waveN*64 + j*16 + lm;
    const float bb = bias ? bias[colG] : 0.f;
    #pragma unroll
    for (int i = 0; i < 2; ++i) {
      #pragma unroll
      for (int r = 0; r < 8; ++r) {
        const int rowG = blockM + waveM*32 + i*16 + mhalf + r;
        const float val = acc[i][j][r] + bb;
        const int bI = rowG >> 10, sI = rowG & (S_ - 1);
        const int hI = colG >> 7,  dI = colG & (HD_ - 1);
        if (dstF) {
          size_t idx = (layoutF == 0) ? ((size_t)rowG*2048 + colG)
                     : (layoutF == 1) ? ((((size_t)bI*H_ + hI)*(size_t)Tdim + sI)*HD_ + dI)
                                      : ((((size_t)bI*H_ + hI)*HD_ + dI)*(size_t)CS_ + sI);
          dstF[idx] = val;
        }
        if (dstB) {
          size_t idx = (layoutB == 0) ? ((size_t)rowG*2048 + colG)
                     : (layoutB == 1) ? ((((size_t)bI*H_ + hI)*(size_t)Tdim + sI)*HD_ + dI)
                                      : ((((size_t)bI*H_ + hI)*HD_ + dI)*(size_t)CS_ + sI);
          dstB[idx] = f32_bf16(val);
        }
      }
    }
  }
}

// ---------------------------------------------------------------------------
// Flash attention: grid (S/64, H, B), 128 threads (4 waves), each wave owns
// 16 q-rows, streams 32 cache positions/step. K (row-major) and V (pre-
// transposed [hd][cs]) staged bf16 via async DMA, double-buffered.
// ---------------------------------------------------------------------------
__global__ __launch_bounds__(128)
void attn_flash(const unsigned short* __restrict__ qbf,   // (B,H,S,HD) bf16
                const unsigned short* __restrict__ kbf,   // (B,H,CS,HD) bf16
                const unsigned short* __restrict__ vbfT,  // (B,H,HD,CS) bf16
                const unsigned char* __restrict__ mask,   // (B,1,S,CS) bool
                unsigned short* __restrict__ attn_out)    // (B,S,H*HD) bf16
{
  __shared__ unsigned short sK[2][32][128];    // [buf][kpos][d]
  __shared__ unsigned short sVt[2][128][32];   // [buf][d][kpos]
  __shared__ unsigned short sP[4][16][32];     // per-wave P staging (C->A relayout)

  const int tid  = threadIdx.x;
  const int lane = tid & 31;
  const int wave = tid >> 5;
  const int b = blockIdx.z, h = blockIdx.y;
  const int qbase = blockIdx.x*64 + wave*16;
  const int lm = lane & 15;
  const int kg = (lane >> 4) * 8;
  const int kh = (lane >> 4) * 16;
  const int mhalf = (lane >> 4) * 8;
  const float scale = 0.08838834764831845f;   // 1/sqrt(128)

  const unsigned sK_base = (unsigned)(size_t)&sK[0][0][0];
  const unsigned sV_base = (unsigned)(size_t)&sVt[0][0][0];
  const size_t kOff = (size_t)(b*H_ + h)*(size_t)CS_*HD_;   // elements
  const size_t vOff = (size_t)(b*H_ + h)*(size_t)HD_*CS_;

  auto stage = [&](int kt, int buf) {
    #pragma unroll
    for (int i = 0; i < 4; ++i) {
      int c = tid + i*128;                  // 32 rows x 16 x 16B
      int row = c >> 4, q16 = c & 15;
      unsigned lds  = sK_base + (unsigned)buf*8192u + (unsigned)row*256u + (unsigned)q16*16u;
      unsigned voff = (unsigned)((kOff + (size_t)(kt + row)*HD_ + q16*8) * 2);
      async_copy16(lds, kbf, voff);
    }
    #pragma unroll
    for (int i = 0; i < 4; ++i) {
      int c = tid + i*128;                  // 128 rows x 4 x 16B
      int row = c >> 2, q16 = c & 3;
      unsigned lds  = sV_base + (unsigned)buf*8192u + (unsigned)row*64u + (unsigned)q16*16u;
      unsigned voff = (unsigned)((vOff + (size_t)row*CS_ + kt + q16*8) * 2);
      async_copy16(lds, vbfT, voff);
    }
  };

  // Q fragments: 4 K-chunks of 32, row = qbase + lm, registers for the whole loop
  Frag16 qf[4];
  const unsigned short* qp = qbf + (((size_t)b*H_ + h)*S_ + qbase + lm)*HD_;
  #pragma unroll
  for (int c = 0; c < 4; ++c) {
    #pragma unroll
    for (int p = 0; p < 8; ++p) {
      int kk = kh + 2*p;
      qf[c].u[2*p]   = qp[c*32 + kk];
      qf[c].u[2*p+1] = qp[c*32 + kk + 1];
    }
  }

  float m_i[8], l_i[8];
  #pragma unroll
  for (int r = 0; r < 8; ++r) { m_i[r] = -1e30f; l_i[r] = 0.f; }
  f32x8 oacc[8];
  #pragma unroll
  for (int d = 0; d < 8; ++d) oacc[d] = zero8();

  const unsigned char* mb = mask + (size_t)b*S_*CS_;

  stage(0, 0);
  for (int kt = 0; kt < CS_; kt += 32) {
    const int cur = (kt >> 5) & 1;
    if (kt + 32 < CS_) {
      stage(kt + 32, cur ^ 1);
      asm volatile("s_wait_asynccnt 0x8" ::: "memory");
    } else {
      asm volatile("s_wait_asynccnt 0x0" ::: "memory");
    }
    __syncthreads();

    // scores: S(16x32) = Q(16x128) @ K_tile^T
    f32x8 sc0 = zero8(), sc1 = zero8();
    #pragma unroll
    for (int dc = 0; dc < 4; ++dc) {
      Frag16 bk0, bk1;
      const unsigned short* p0 = &sK[cur][lm][dc*32];
      const unsigned short* p1 = &sK[cur][16 + lm][dc*32];
      #pragma unroll
      for (int p = 0; p < 8; ++p) {
        bk0.u[2*p] = p0[kh + 2*p]; bk0.u[2*p+1] = p0[kh + 2*p + 1];
        bk1.u[2*p] = p1[kh + 2*p]; bk1.u[2*p+1] = p1[kh + 2*p + 1];
      }
      sc0 = __builtin_amdgcn_wmma_f32_16x16x32_bf16(false, qf[dc].v, false, bk0.v, (short)0, sc0, false, false);
      sc1 = __builtin_amdgcn_wmma_f32_16x16x32_bf16(false, qf[dc].v, false, bk1.v, (short)0, sc1, false, false);
    }

    // mask + scale + online softmax update
    float corr[8];
    #pragma unroll
    for (int r = 0; r < 8; ++r) {
      const int q = qbase + mhalf + r;
      const unsigned char* mq = mb + (size_t)q*CS_ + kt;
      float s0 = mq[lm]      ? sc0[r]*scale : -1e30f;
      float s1 = mq[16 + lm] ? sc1[r]*scale : -1e30f;
      float rm = fmaxf(s0, s1);
      #pragma unroll
      for (int off = 1; off < 16; off <<= 1) rm = fmaxf(rm, __shfl_xor(rm, off, 32));
      float mnew = fmaxf(m_i[r], rm);
      corr[r] = __expf(m_i[r] - mnew);
      float p0 = __expf(s0 - mnew);
      float p1 = __expf(s1 - mnew);
      float rs = p0 + p1;
      #pragma unroll
      for (int off = 1; off < 16; off <<= 1) rs += __shfl_xor(rs, off, 32);
      l_i[r] = l_i[r]*corr[r] + rs;
      m_i[r] = mnew;
      sP[wave][mhalf + r][lm]      = f32_bf16(p0);
      sP[wave][mhalf + r][16 + lm] = f32_bf16(p1);
    }
    #pragma unroll
    for (int d = 0; d < 8; ++d)
      #pragma unroll
      for (int r = 0; r < 8; ++r)
        oacc[d][r] *= corr[r];

    // wave-private LDS RAW: make P stores visible before fragment reload
    asm volatile("s_wait_dscnt 0" ::: "memory");

    Frag16 pf;
    const unsigned short* pp = &sP[wave][lm][0];
    #pragma unroll
    for (int p = 0; p < 8; ++p) {
      int kk = kh + 2*p;
      pf.u[2*p]   = pp[kk];
      pf.u[2*p+1] = pp[kk+1];
    }
    // O(16x128) += P(16x32) @ V(32x128)
    #pragma unroll
    for (int d = 0; d < 8; ++d) {
      Frag16 vf;
      const unsigned short* vp = &sVt[cur][d*16 + lm][0];
      #pragma unroll
      for (int p = 0; p < 8; ++p) {
        vf.u[2*p]   = vp[kh + 2*p];
        vf.u[2*p+1] = vp[kh + 2*p + 1];
      }
      oacc[d] = __builtin_amdgcn_wmma_f32_16x16x32_bf16(false, pf.v, false, vf.v, (short)0, oacc[d], false, false);
    }
    __syncthreads();   // protect read buffer from next iteration's DMA writes
  }

  float inv[8];
  #pragma unroll
  for (int r = 0; r < 8; ++r) inv[r] = 1.f / l_i[r];
  #pragma unroll
  for (int d = 0; d < 8; ++d) {
    const int dcol = d*16 + lm;
    #pragma unroll
    for (int r = 0; r < 8; ++r) {
      const int q = qbase + mhalf + r;
      attn_out[((size_t)b*S_ + q)*(size_t)NQK + h*HD_ + dcol] = f32_bf16(oacc[d][r]*inv[r]);
    }
  }
}

// ---------------------------------------------------------------------------
// Copy cache positions [S, CS) (contiguous per (b,h)) into upd_k / upd_v
// ---------------------------------------------------------------------------
__global__ void copy_cache_tail(const float* __restrict__ ck, const float* __restrict__ cv,
                                float* __restrict__ uk, float* __restrict__ uv)
{
  size_t idx = (size_t)blockIdx.x * blockDim.x + threadIdx.x;
  const size_t seglen = (size_t)(CS_ - S_) * HD_;
  const size_t total  = (size_t)B_ * H_ * seglen;
  if (idx >= total) return;
  size_t seg = idx / seglen, off = idx - seg*seglen;
  size_t o = seg*(size_t)CS_*HD_ + (size_t)S_*HD_ + off;
  uk[o] = ck[o];
  uv[o] = cv[o];
}

// ---------------------------------------------------------------------------
extern "C" void kernel_launch(void* const* d_in, const int* in_sizes, int n_in,
                              void* d_out, int out_size, void* d_ws, size_t ws_size,
                              hipStream_t stream) {
  (void)in_sizes; (void)n_in; (void)out_size; (void)ws_size;
  const float* x  = (const float*)d_in[0];
  const float* Wq = (const float*)d_in[1];
  const float* bq = (const float*)d_in[2];
  const float* Wk = (const float*)d_in[3];
  const float* Wv = (const float*)d_in[4];
  const float* bv = (const float*)d_in[5];
  const float* Wo = (const float*)d_in[6];
  const float* bo = (const float*)d_in[7];
  const float* ck = (const float*)d_in[8];
  const float* cv = (const float*)d_in[9];
  const unsigned char* mask = (const unsigned char*)d_in[10];

  float* out   = (float*)d_out;                         // (B,S,D)
  float* upd_k = out + (size_t)B_*S_*D_;                // (B,H,CS,HD)
  float* upd_v = upd_k + (size_t)B_*H_*CS_*HD_;         // (B,H,CS,HD)

  char* w = (char*)d_ws;                                // bf16 scratch (~88 MB)
  unsigned short* xbf    = (unsigned short*)(w);                       // 8 MB
  unsigned short* Wqt    = (unsigned short*)(w + ((size_t)8  << 20));  // 8 MB
  unsigned short* Wkt    = (unsigned short*)(w + ((size_t)16 << 20));  // 8 MB
  unsigned short* Wvt    = (unsigned short*)(w + ((size_t)24 << 20));  // 8 MB
  unsigned short* Wot    = (unsigned short*)(w + ((size_t)32 << 20));  // 8 MB
  unsigned short* qbf    = (unsigned short*)(w + ((size_t)40 << 20));  // 8 MB
  unsigned short* kbf    = (unsigned short*)(w + ((size_t)48 << 20));  // 16 MB
  unsigned short* vbfT   = (unsigned short*)(w + ((size_t)64 << 20));  // 16 MB
  unsigned short* attnbf = (unsigned short*)(w + ((size_t)80 << 20));  // 8 MB

  // one-time conversions
  const int nx = B_*S_*D_;
  f32_to_bf16_vec<<<dim3((nx + 255)/256), 256, 0, stream>>>(x, xbf, nx);
  dim3 tgrid(64, 64);
  transpose_f32_to_bf16<<<tgrid, 256, 0, stream>>>(Wq, Wqt);
  transpose_f32_to_bf16<<<tgrid, 256, 0, stream>>>(Wk, Wkt);
  transpose_f32_to_bf16<<<tgrid, 256, 0, stream>>>(Wv, Wvt);
  transpose_f32_to_bf16<<<tgrid, 256, 0, stream>>>(Wo, Wot);

  dim3 ggrid(16, 16);   // 2048/128 x 2048/128
  // QKV projections: K/V dual-store (f32 cache in d_out + bf16 for attention)
  gemm_bf16<<<ggrid, 256, 0, stream>>>(xbf, Wqt, bq,      nullptr, 0, qbf,  1, S_);
  gemm_bf16<<<ggrid, 256, 0, stream>>>(xbf, Wkt, nullptr, upd_k,   1, kbf,  1, CS_);
  gemm_bf16<<<ggrid, 256, 0, stream>>>(xbf, Wvt, bv,      upd_v,   1, vbfT, 2, CS_);

  {
    size_t total = (size_t)B_*H_*(CS_ - S_)*HD_;
    copy_cache_tail<<<dim3((unsigned)((total + 255)/256)), 256, 0, stream>>>(ck, cv, upd_k, upd_v);
  }

  attn_flash<<<dim3(S_/64, H_, B_), 128, 0, stream>>>(qbf, kbf, vbfT, mask, attnbf);

  // output projection
  gemm_bf16<<<ggrid, 256, 0, stream>>>(attnbf, Wot, bo, out, 0, nullptr, 0, 0);
}
